// CustomMHA_87935160419038
// MI455X (gfx1250) — matrix operand order repaired and checked by reading, not compiled, not run on hardware
//
#include <hip/hip_runtime.h>
#include <hip/hip_bf16.h>
#include <stdint.h>

// Problem constants (match reference)
#define B_   4
#define H_   16
#define L_   2048
#define DK_  64
#define D_   1024
#define M_   (B_ * L_)        // 8192 rows for both GEMMs
#define SCALE_ 0.125f         // 1/sqrt(64)

typedef __attribute__((ext_vector_type(16))) __bf16 v16bf;
typedef __attribute__((ext_vector_type(8)))  float  v8f;
typedef __attribute__((ext_vector_type(4)))  unsigned int v4u;
typedef __attribute__((ext_vector_type(8)))  int v8i;
typedef __attribute__((ext_vector_type(4)))  int v4i;

union Frag {                  // 16 bf16 = 32 bytes, WMMA A/B operand
  v16bf f;
  uint4 q[2];
};

static __device__ __forceinline__ unsigned short f2bf(float x) {
  union { float f; unsigned u; } c; c.f = x;
  unsigned r = c.u + 0x7FFFu + ((c.u >> 16) & 1u);   // round-to-nearest-even
  return (unsigned short)(r >> 16);
}

static __device__ __forceinline__ v8f wmma_bf16(Frag a, Frag b, v8f c) {
  // (neg_a, A, neg_b, B, c_mod, C, reuse_a, reuse_b)
  return __builtin_amdgcn_wmma_f32_16x16x32_bf16(false, a.f, false, b.f,
                                                 (short)0, c, false, false);
}

// ds_swizzle xor-shuffle within 16-lane groups (no LDS memory touched)
template <int MASK>
static __device__ __forceinline__ float swz_xor(float v) {
  int r = __builtin_amdgcn_ds_swizzle(__float_as_int(v), (MASK << 10) | 0x1F);
  return __int_as_float(r);
}

// ---------------------------------------------------------------------------
// Tensor Data Mover: 2D bf16 tile, global -> LDS.
// D# per ISA ch.8: group0 = {count|flags, lds_addr, gaddr_lo, gaddr_hi|type=2}
// group1 packs data_size/pad + tensor dims (== tile dims; tiles never OOB),
// tile dims and dim0 stride. pad = 4 dwords (8 halfs) every 2^(code+1) dwords,
// reproducing the 16B-aligned padded LDS row strides the frag loads assume.
// This toolchain's builtin is the 6-arg form (extra int32x8 group).
// ---------------------------------------------------------------------------
static __device__ __forceinline__ void tdm_load_bf16_2d(
    const void* gptr, unsigned lds_off,
    unsigned pitch_elems,     // tensor_dim0_stride (elements)
    unsigned tile_d0,         // contiguous elements per row
    unsigned tile_d1,         // rows
    int pad_interval_code)    // -1 = no padding
{
  unsigned long long ga = (unsigned long long)(uintptr_t)gptr;
  v4u g0;
  g0[0] = 1u;                                    // count = 1 (valid, user mode)
  g0[1] = lds_off;                               // LDS byte address
  g0[2] = (unsigned)ga;                          // global addr [31:0]
  g0[3] = (unsigned)((ga >> 32) & 0x01FFFFFFu)   // global addr [56:32]
          | (2u << 30);                          // type = 2 (image)
  unsigned flags = 1u << 16;                     // data_size = 1 -> 2 bytes
  if (pad_interval_code >= 0)
    flags |= (1u << 20)                          // pad_enable
          |  ((unsigned)pad_interval_code << 22) // pad interval
          |  (3u << 25);                         // pad amount = 4 dwords
  v8i g1;
  g1[0] = (int)flags;
  g1[1] = (int)((tile_d0 & 0xFFFFu) << 16);      // tensor_dim0 lo16 (hi = atomic addr = 0)
  g1[2] = (int)(((tile_d0 >> 16) & 0xFFFFu) | ((tile_d1 & 0xFFFFu) << 16)); // dim0 hi | dim1 lo
  g1[3] = (int)(((tile_d1 >> 16) & 0xFFFFu) | ((tile_d0 & 0xFFFFu) << 16)); // dim1 hi | tile_dim0
  g1[4] = (int)(tile_d1 & 0xFFFFu);              // tile_dim1 (tile_dim2 = 0)
  g1[5] = (int)pitch_elems;                      // tensor_dim0_stride lo32
  g1[6] = 0;
  g1[7] = 0;
  v4i z4 = {0, 0, 0, 0};
  v8i z8 = {0, 0, 0, 0, 0, 0, 0, 0};
  __builtin_amdgcn_tensor_load_to_lds(g0, g1, z4, z4, z8, 0);
}

static __device__ __forceinline__ unsigned lds_off_of(const void* p) {
  return (unsigned)(uintptr_t)p;                 // flat LDS addr: offset in [31:0]
}

// ---------------------------------------------------------------------------
// Kernel 0: fp32 -> bf16 (vectorized x4)
// ---------------------------------------------------------------------------
__global__ __launch_bounds__(256) void cvt_f32_bf16(const float* __restrict__ src,
                                                    unsigned short* __restrict__ dst,
                                                    int n4) {
  int i = blockIdx.x * blockDim.x + threadIdx.x;
  if (i >= n4) return;
  float4 v = ((const float4*)src)[i];
  unsigned lo = (unsigned)f2bf(v.x) | ((unsigned)f2bf(v.y) << 16);
  unsigned hi = (unsigned)f2bf(v.z) | ((unsigned)f2bf(v.w) << 16);
  uint2 t; t.x = lo; t.y = hi;
  ((uint2*)dst)[i] = t;
}

// ---------------------------------------------------------------------------
// Kernels 1 & 3: C[m,n] = sum_d A[m,d]*W[n,d] + bias[n]
//   256(M) x 64(N) tile per 256-thread block (8 waves). Each wave: 2 M-subtiles
//   x 4 N-subtiles = 8 WMMAs per 32-deep K step. Tiles streamed into
//   double-buffered LDS by the Tensor Data Mover (wave 0 issues, TENSORcnt).
//   All 4 B frags are fetched before the WMMA burst to batch ds_loads.
// MODE 0: scatter bf16 into qkv (q,k: [B][H][L][64]; v transposed [B][H][64][L])
// MODE 1: fp32 out [M][1024]
// ---------------------------------------------------------------------------
template <int MODE>
__global__ __launch_bounds__(256) void gemm_bf16_k(
    const unsigned short* __restrict__ A,
    const unsigned short* __restrict__ W,
    const float* __restrict__ bias,
    unsigned short* __restrict__ qkv,
    float* __restrict__ fout) {
  __shared__ unsigned short As[2][256][40];   // 32 cols + 8 pad -> 80B stride
  __shared__ unsigned short Bs[2][64][40];

  const int tid  = threadIdx.x;
  const int wave = tid >> 5;
  const int lane = tid & 31;
  const int lr   = lane & 15;
  const int lh   = lane >> 4;
  const int m0   = blockIdx.y * 256;
  const int n0   = blockIdx.x * 64;

  v8f acc[8] = {};

  // Prologue: TDM-load first K slab into buffer 0
  if (wave == 0) {
    tdm_load_bf16_2d(A + (size_t)m0 * D_, lds_off_of(&As[0][0][0]),
                     D_, 32, 256, /*pad: 16dw interval*/ 3);
    tdm_load_bf16_2d(W + (size_t)n0 * D_, lds_off_of(&Bs[0][0][0]),
                     D_, 32, 64, 3);
  }

  int cur = 0;
  for (int k0 = 0; k0 < D_; k0 += 32) {
    if (wave == 0) __builtin_amdgcn_s_wait_tensorcnt(0);
    __syncthreads();   // buf[cur] ready for all waves; buf[cur^1] free to fill
    if (wave == 0 && k0 + 32 < D_) {
      tdm_load_bf16_2d(A + (size_t)m0 * D_ + (k0 + 32), lds_off_of(&As[cur ^ 1][0][0]),
                       D_, 32, 256, 3);
      tdm_load_bf16_2d(W + (size_t)n0 * D_ + (k0 + 32), lds_off_of(&Bs[cur ^ 1][0][0]),
                       D_, 32, 64, 3);
    }

    // Batch all fragment loads, then burst the WMMAs.
    Frag a0, a1, b[4];
    {   // A frags: rows wave*32 + lr and +16; K split per 16-bit A layout
      const unsigned short* r0 = &As[cur][wave * 32 + lr][0];
      const unsigned short* r1 = &As[cur][wave * 32 + 16 + lr][0];
      a0.q[0] = *(const uint4*)(r0 + lh * 8);
      a0.q[1] = *(const uint4*)(r0 + 16 + lh * 8);
      a1.q[0] = *(const uint4*)(r1 + lh * 8);
      a1.q[1] = *(const uint4*)(r1 + 16 + lh * 8);
    }
#pragma unroll
    for (int n = 0; n < 4; ++n) {
      const unsigned short* brow = &Bs[cur][n * 16 + lr][0];
      b[n].q[0] = *(const uint4*)(brow + lh * 8);
      b[n].q[1] = *(const uint4*)(brow + 16 + lh * 8);
    }
#pragma unroll
    for (int n = 0; n < 4; ++n) {
      acc[n]     = wmma_bf16(a0, b[n], acc[n]);
      acc[4 + n] = wmma_bf16(a1, b[n], acc[4 + n]);
    }
    cur ^= 1;
  }

  // Epilogue. acc[half*4+n][r] -> row m0+wave*32+half*16+r+8*lh, col n0+n*16+lr
#pragma unroll
  for (int half = 0; half < 2; ++half) {
#pragma unroll
    for (int n = 0; n < 4; ++n) {
      int col = n0 + n * 16 + lr;
      float bv = bias[col];
#pragma unroll
      for (int r = 0; r < 8; ++r) {
        int row = m0 + wave * 32 + half * 16 + r + lh * 8;
        float v = acc[half * 4 + n][r] + bv;
        if (MODE == 0) {
          int t  = col >> 10;            // 0=q 1=k 2=v
          int hh = (col & 1023) >> 6;
          int dk = col & 63;
          int bb = row >> 11;            // batch
          int ll = row & 2047;           // seq pos
          size_t head = ((size_t)t * B_ + bb) * H_ + hh;
          size_t idx = (t == 2) ? (head * DK_ + dk) * L_ + ll   // V transposed
                                : (head * L_ + ll) * DK_ + dk;  // Q, K row-major
          qkv[idx] = f2bf(v);
        } else {
          fout[(size_t)row * D_ + col] = v;
        }
      }
    }
  }
}

// ---------------------------------------------------------------------------
// Kernel 2: flash attention. Block = 128 threads (4 waves); each wave owns a
// 16-query tile. K [bh][l][dk] and pre-transposed V [bh][dk][l] tiles are
// streamed into double-buffered LDS by the TDM; softmax row reductions use
// ds_swizzle xor patterns; P relayout goes through per-wave LDS scratch.
// B fragments are batch-loaded before each WMMA burst.
// ---------------------------------------------------------------------------
__global__ __launch_bounds__(128) void attn_k(
    const unsigned short* __restrict__ Q,   // [B][H][L][64] bf16
    const unsigned short* __restrict__ K,   // [B][H][L][64] bf16
    const unsigned short* __restrict__ V,   // [B][H][64][L] bf16 (transposed)
    unsigned short* __restrict__ Z) {       // [B][L][H*64] bf16
  __shared__ unsigned short Ks[2][32][72];  // keys x d (pad -> 144B stride)
  __shared__ unsigned short Vt[2][64][40];  // d x keys (pad -> 80B stride)
  __shared__ unsigned short Ps[4][16][40];  // per-wave P relayout scratch

  const int tid  = threadIdx.x;
  const int wave = tid >> 5;
  const int lane = tid & 31;
  const int lr   = lane & 15;
  const int lh   = lane >> 4;
  const int bh   = blockIdx.y;              // b*H + h
  const int q0   = blockIdx.x * 64;
  const size_t baseq = (size_t)bh * L_ * DK_;   // Q/K base for this head
  const size_t basev = (size_t)bh * DK_ * L_;   // Vt base for this head

  // Q fragments (A operand), loaded once: rows q0+wave*16+(lane&15)
  Frag qa0, qa1;
  {
    const unsigned short* qrow = Q + baseq + (size_t)(q0 + wave * 16 + lr) * DK_;
    qa0.q[0] = *(const uint4*)(qrow + lh * 8);
    qa0.q[1] = *(const uint4*)(qrow + 16 + lh * 8);
    qa1.q[0] = *(const uint4*)(qrow + 32 + lh * 8);
    qa1.q[1] = *(const uint4*)(qrow + 48 + lh * 8);
  }

  float mrow[8], lrow[8];
  v8f o[4] = {};
#pragma unroll
  for (int r = 0; r < 8; ++r) { mrow[r] = -3.0e38f; lrow[r] = 0.0f; }

  // Prologue TDM: first 32-key tiles into buffer 0
  if (wave == 0) {
    tdm_load_bf16_2d(K + baseq, lds_off_of(&Ks[0][0][0]), DK_, DK_, 32, /*32dw*/ 4);
    tdm_load_bf16_2d(V + basev, lds_off_of(&Vt[0][0][0]), L_, 32, DK_, /*16dw*/ 3);
  }

  int cur = 0;
  for (int k0 = 0; k0 < L_; k0 += 32) {
    if (wave == 0) __builtin_amdgcn_s_wait_tensorcnt(0);
    __syncthreads();
    if (wave == 0 && k0 + 32 < L_) {
      tdm_load_bf16_2d(K + baseq + (size_t)(k0 + 32) * DK_,
                       lds_off_of(&Ks[cur ^ 1][0][0]), DK_, DK_, 32, 4);
      tdm_load_bf16_2d(V + basev + (k0 + 32),
                       lds_off_of(&Vt[cur ^ 1][0][0]), L_, 32, DK_, 3);
    }

    // S = Q(16x64) @ K^T(64x32): batch 4 B frags (8 ds_loads), then 4 WMMAs
    Frag kb[4];
    kb[0].q[0] = *(const uint4*)(&Ks[cur][lr][0] + lh * 8);
    kb[0].q[1] = *(const uint4*)(&Ks[cur][lr][0] + 16 + lh * 8);
    kb[1].q[0] = *(const uint4*)(&Ks[cur][lr][32] + lh * 8);
    kb[1].q[1] = *(const uint4*)(&Ks[cur][lr][32] + 16 + lh * 8);
    kb[2].q[0] = *(const uint4*)(&Ks[cur][16 + lr][0] + lh * 8);
    kb[2].q[1] = *(const uint4*)(&Ks[cur][16 + lr][0] + 16 + lh * 8);
    kb[3].q[0] = *(const uint4*)(&Ks[cur][16 + lr][32] + lh * 8);
    kb[3].q[1] = *(const uint4*)(&Ks[cur][16 + lr][32] + 16 + lh * 8);
    v8f s0 = {}, s1 = {};
    s0 = wmma_bf16(qa0, kb[0], s0);
    s0 = wmma_bf16(qa1, kb[1], s0);
    s1 = wmma_bf16(qa0, kb[2], s1);
    s1 = wmma_bf16(qa1, kb[3], s1);

    // Online softmax. C layout: lane holds row r+8*lh, col lr (s0: keys 0-15,
    // s1: keys 16-31). Row reductions across 16 lanes via ds_swizzle xor.
#pragma unroll
    for (int r = 0; r < 8; ++r) {
      float a = s0[r] * SCALE_;
      float b = s1[r] * SCALE_;
      float rm = fmaxf(a, b);
      rm = fmaxf(rm, swz_xor<8>(rm));
      rm = fmaxf(rm, swz_xor<4>(rm));
      rm = fmaxf(rm, swz_xor<2>(rm));
      rm = fmaxf(rm, swz_xor<1>(rm));
      float mnew  = fmaxf(mrow[r], rm);
      float alpha = __expf(mrow[r] - mnew);
      float p0 = __expf(a - mnew);
      float p1 = __expf(b - mnew);
      float rs = p0 + p1;
      rs += swz_xor<8>(rs);
      rs += swz_xor<4>(rs);
      rs += swz_xor<2>(rs);
      rs += swz_xor<1>(rs);
      lrow[r] = lrow[r] * alpha + rs;
      mrow[r] = mnew;
      o[0][r] = o[0][r] * alpha;
      o[1][r] = o[1][r] * alpha;
      o[2][r] = o[2][r] * alpha;
      o[3][r] = o[3][r] * alpha;
      int prow = r + lh * 8;
      Ps[wave][prow][lr]      = f2bf(p0);
      Ps[wave][prow][16 + lr] = f2bf(p1);
    }
    // Per-wave LDS relayout: DS ops are in-order per wave; drain stores and
    // keep the compiler from reordering the fragment reloads above them.
    asm volatile("s_wait_dscnt 0" ::: "memory");

    // O += P(16x32) @ V(32x64): batch P + 4 V frags, then 4 WMMAs
    Frag pf, vb[4];
    {
      const unsigned short* prow = &Ps[wave][lr][0];
      pf.q[0] = *(const uint4*)(prow + lh * 8);
      pf.q[1] = *(const uint4*)(prow + 16 + lh * 8);
    }
#pragma unroll
    for (int n = 0; n < 4; ++n) {
      const unsigned short* vrow = &Vt[cur][n * 16 + lr][0];
      vb[n].q[0] = *(const uint4*)(vrow + lh * 8);
      vb[n].q[1] = *(const uint4*)(vrow + 16 + lh * 8);
    }
#pragma unroll
    for (int n = 0; n < 4; ++n)
      o[n] = wmma_bf16(pf, vb[n], o[n]);
    cur ^= 1;
  }

  // Normalize and emit z in [B][L][D] (head-interleaved) bf16
  const int bb = bh >> 4;
  const int hh = bh & 15;
#pragma unroll
  for (int r = 0; r < 8; ++r) {
    float inv = 1.0f / lrow[r];
    int ll = q0 + wave * 16 + r + lh * 8;
    unsigned short* zrow = Z + ((size_t)bb * L_ + ll) * D_ + hh * DK_;
#pragma unroll
    for (int n = 0; n < 4; ++n)
      zrow[n * 16 + lr] = f2bf(o[n][r] * inv);
  }
}

// ---------------------------------------------------------------------------
// Host launcher
// ---------------------------------------------------------------------------
extern "C" void kernel_launch(void* const* d_in, const int* in_sizes, int n_in,
                              void* d_out, int out_size, void* d_ws, size_t ws_size,
                              hipStream_t stream) {
  (void)in_sizes; (void)n_in; (void)out_size; (void)ws_size;
  const float* x      = (const float*)d_in[0];   // [B,L,D]
  const float* Win_w  = (const float*)d_in[1];   // [3D,D]
  const float* Win_b  = (const float*)d_in[2];   // [3D]
  const float* proj_w = (const float*)d_in[3];   // [D,D]
  const float* proj_b = (const float*)d_in[4];   // [D]
  float* out = (float*)d_out;                    // [B,L,D] fp32

  // Workspace carve-up (256B aligned)
  char* ws = (char*)d_ws;
  size_t off = 0;
  auto carve = [&](size_t bytes) -> char* {
    off = (off + 255) & ~(size_t)255;
    char* p = ws + off;
    off += bytes;
    return p;
  };
  unsigned short* XB  = (unsigned short*)carve((size_t)M_ * D_ * 2);       // x bf16
  unsigned short* WB  = (unsigned short*)carve((size_t)3 * D_ * D_ * 2);   // Win bf16
  unsigned short* PB  = (unsigned short*)carve((size_t)D_ * D_ * 2);       // proj bf16
  unsigned short* QKV = (unsigned short*)carve((size_t)3 * B_ * H_ * L_ * DK_ * 2);
  unsigned short* ZB  = (unsigned short*)carve((size_t)M_ * D_ * 2);       // attn out bf16

  const unsigned short* Qb = QKV;
  const unsigned short* Kb = QKV + (size_t)B_ * H_ * L_ * DK_;
  const unsigned short* Vb = QKV + (size_t)2 * B_ * H_ * L_ * DK_;

  // 0) fp32 -> bf16 conversions
  {
    int n4 = (M_ * D_) / 4;
    cvt_f32_bf16<<<n4 / 256, 256, 0, stream>>>(x, XB, n4);
  }
  {
    int n4 = (3 * D_ * D_) / 4;
    cvt_f32_bf16<<<n4 / 256, 256, 0, stream>>>(Win_w, WB, n4);
  }
  {
    int n4 = (D_ * D_) / 4;
    cvt_f32_bf16<<<n4 / 256, 256, 0, stream>>>(proj_w, PB, n4);
  }

  // 1) QKV projection: [8192,1024] x [3072,1024]^T + b -> q/k/v bf16
  {
    dim3 grid(3 * D_ / 64, M_ / 256);
    gemm_bf16_k<0><<<grid, 256, 0, stream>>>(XB, WB, Win_b, QKV, (float*)nullptr);
  }

  // 2) attention
  {
    dim3 grid(L_ / 64, B_ * H_);
    attn_k<<<grid, 128, 0, stream>>>(Qb, Kb, Vb, ZB);
  }

  // 3) output projection: [8192,1024] x [1024,1024]^T + b -> fp32 out
  {
    dim3 grid(D_ / 64, M_ / 256);
    gemm_bf16_k<1><<<grid, 256, 0, stream>>>(ZB, PB, proj_b,
                                             (unsigned short*)nullptr, out);
  }
}
